// BinaryTreeLSTM_86904368267877
// MI455X (gfx1250) — matrix operand order repaired and checked by reading, not compiled, not run on hardware
//
#include <hip/hip_runtime.h>
#include <cstdint>
#include <cstddef>

// ---------------------------------------------------------------------------
// BinaryTreeLSTM on gfx1250 (MI455X).
//  - bf16 WMMA GEMMs, fp32 accumulate, zero-padded K so GEMM loops have no
//    guarded loads or divergent branches (EXEC all-ones around every WMMA).
//  - Leaf kernel converts fp32 X -> bf16 in registers (no pre-pass).
//  - Last 6 tree levels run in one persistent workgroup through LDS.
// ---------------------------------------------------------------------------

typedef __attribute__((ext_vector_type(16))) __bf16       v16bf;
typedef __attribute__((ext_vector_type(8)))  float        v8f;
typedef __attribute__((ext_vector_type(4)))  float        v4f;
typedef __attribute__((ext_vector_type(8)))  unsigned int v8u;
typedef __attribute__((ext_vector_type(4)))  unsigned int v4u;

#define N_LEAVES 131072
#define IN_DIM   300
#define MEM_DIM  150
#define K_XPAD   320      // IN_DIM padded to mult of 32 (WlP row stride)
#define N_PAD    160      // MEM_DIM padded to mult of 32 (c/h row stride)
#define ROW_SLACK 32
#define TAIL_M   32       // first level handled by the persistent tail kernel

__device__ __forceinline__ float sigf(float x) {
    return 1.0f / (1.0f + __expf(-x));
}

// ---------------------------------------------------------------------------
// Prep: pad+convert weights/biases (tiny, one-shot).
// ---------------------------------------------------------------------------
__global__ void prep_weights_kernel(const float* __restrict__ Wl,
                                    const float* __restrict__ bl,
                                    const float* __restrict__ Wi,
                                    const float* __restrict__ bi,
                                    __bf16* __restrict__ WlP,   // (160,320)
                                    float*  __restrict__ blP,   // (160,)
                                    __bf16* __restrict__ WiP,   // (5,160,160)
                                    float*  __restrict__ biP) { // (5,160)
    int i = blockIdx.x * blockDim.x + threadIdx.x;
    if (i < N_PAD * K_XPAD) {
        int r = i / K_XPAD, k = i % K_XPAD;
        WlP[i] = (__bf16)((r < MEM_DIM && k < IN_DIM) ? Wl[r * IN_DIM + k] : 0.0f);
    }
    if (i < 5 * N_PAD * N_PAD) {
        int g = i / (N_PAD * N_PAD);
        int rem = i % (N_PAD * N_PAD);
        int r = rem / N_PAD, k = rem % N_PAD;
        WiP[i] = (__bf16)((r < MEM_DIM && k < MEM_DIM)
                              ? Wi[(g * MEM_DIM + r) * MEM_DIM + k] : 0.0f);
    }
    if (i < N_PAD) blP[i] = (i < MEM_DIM) ? bl[i] : 0.0f;
    if (i < 5 * N_PAD) {
        int g = i / N_PAD, n = i % N_PAD;
        biP[i] = (n < MEM_DIM) ? bi[g * MEM_DIM + n] : 0.0f;
    }
}

// ---------------------------------------------------------------------------
// Fragment helpers (gfx1250 wave32 16-bit WMMA layouts).
// ---------------------------------------------------------------------------
__device__ __forceinline__ v16bf load_b_bf16(const __bf16* p /*16B aligned*/) {
    const v4u lo = *(const v4u*)p;          // K = khB .. khB+7
    const v4u hi = *(const v4u*)(p + 16);   // K = khB+16 .. khB+23
    v8u u = {lo.x, lo.y, lo.z, lo.w, hi.x, hi.y, hi.z, hi.w};
    return __builtin_bit_cast(v16bf, u);
}

// Build bf16 A fragment from 16 fp32 values in four v4f chunks.
__device__ __forceinline__ v16bf cvt_a_frag(v4f x0, v4f x1, v4f x2, v4f x3) {
    v16bf a;
    #pragma unroll
    for (int e = 0; e < 4; ++e) {
        a[e]      = (__bf16)x0[e];
        a[4 + e]  = (__bf16)x1[e];
        a[8 + e]  = (__bf16)x2[e];
        a[12 + e] = (__bf16)x3[e];
    }
    return a;
}

// A fragment = (hl + hr) rows, fp32 source with zero-padded columns.
__device__ __forceinline__ v16bf build_a_sum(const float* hl, const float* hr,
                                             int off /* = kb + khA */) {
    const v4f s0 = *(const v4f*)(hl + off)      + *(const v4f*)(hr + off);
    const v4f s1 = *(const v4f*)(hl + off + 4)  + *(const v4f*)(hr + off + 4);
    const v4f s2 = *(const v4f*)(hl + off + 16) + *(const v4f*)(hr + off + 16);
    const v4f s3 = *(const v4f*)(hl + off + 20) + *(const v4f*)(hr + off + 20);
    return cvt_a_frag(s0, s1, s2, s3);
}

// ---------------------------------------------------------------------------
// Leaf stage: c = X @ Wl^T + bl ; h = sigmoid(c)*tanh(c).
// fp32 X read directly (once), converted to bf16 in registers.
// One wave: 16(M) x 160(N) strip, 10 accumulators, 100 v_wmma.
// K tail (k=288..299): pointer-clamped loads; garbage K rows hit WlP's
// zero-padded B rows so they contribute exactly 0.
// ---------------------------------------------------------------------------
__global__ void __launch_bounds__(256)
leaf_kernel(const float*  __restrict__ X,    // (131072, 300) fp32
            const __bf16* __restrict__ WlP,  // (160, 320) bf16 zero-padded
            const float*  __restrict__ blP,  // (160,)
            float* __restrict__ cOut,        // (131072+, 160)
            float* __restrict__ hOut) {
    const int lane = threadIdx.x & 31;
    const int wave = threadIdx.x >> 5;
    const int m0   = (blockIdx.x * 8 + wave) * 16;
    const int l15  = lane & 15;
    const int half = lane >> 4;
    const int khA  = half * 8;
    const int khB  = half * 8;

    const float* __restrict__ xrow = X + (size_t)(m0 + l15) * IN_DIM;

    v8f acc[10];
    #pragma unroll
    for (int t = 0; t < 10; ++t) acc[t] = (v8f){};

    // 9 fully in-bounds K steps: k = 0..287.
    for (int kb = 0; kb < 288; kb += 32) {
        const float* p = xrow + kb + khA;
        const v16bf a = cvt_a_frag(*(const v4f*)p,        *(const v4f*)(p + 4),
                                   *(const v4f*)(p + 16), *(const v4f*)(p + 20));
        #pragma unroll
        for (int t = 0; t < 10; ++t) {
            const int col = t * 16 + l15;
            const v16bf b = load_b_bf16(WlP + (size_t)col * K_XPAD + kb + khB);
            acc[t] = __builtin_amdgcn_wmma_f32_16x16x32_bf16(
                         false, a, false, b, (short)0, acc[t], false, false);
        }
    }
    // Tail K step kb=288 (valid k 288..299; k>=300 hits zero B rows).
    {
        const float* p1 = xrow + 288 + khA;              // k 288+khA..291+khA (<300)
        const float* p2 = (khA == 0) ? (xrow + 292)      // k 292..295 (real)
                                     : xrow;             // clamped: values hit B==0
        v16bf a = cvt_a_frag(*(const v4f*)p1, *(const v4f*)p2,
                             (v4f){}, (v4f){});          // k>=304 rows: B==0
        #pragma unroll
        for (int t = 0; t < 10; ++t) {
            const int col = t * 16 + l15;
            const v16bf b = load_b_bf16(WlP + (size_t)col * K_XPAD + 288 + khB);
            acc[t] = __builtin_amdgcn_wmma_f32_16x16x32_bf16(
                         false, a, false, b, (short)0, acc[t], false, false);
        }
    }

    #pragma unroll
    for (int t = 0; t < 10; ++t) {
        const int  col   = t * 16 + l15;
        const bool colOK = (col < MEM_DIM);
        const float bias = blP[col];
        #pragma unroll
        for (int r = 0; r < 8; ++r) {
            const int m = m0 + r + (half << 3);
            const float cc = acc[t][r] + bias;
            const float hh = sigf(cc) * tanhf(cc);
            // Unconditional stores; pad columns forced to 0 (they feed the
            // next level's padded-K GEMM).
            cOut[(size_t)m * N_PAD + col] = colOK ? cc : 0.0f;
            hOut[(size_t)m * N_PAD + col] = colOK ? hh : 0.0f;
        }
    }
}

// ---------------------------------------------------------------------------
// One big tree level (grid-wide): 16(M) x 16(j) x 5 gates per wave, 25 wmma.
// ---------------------------------------------------------------------------
__global__ void __launch_bounds__(256)
tree_level_kernel(const float* __restrict__ cPrev,   // (2M+, 160)
                  const float* __restrict__ hPrev,   // (2M+, 160), pads zero
                  const __bf16* __restrict__ WiP,    // (5,160,160) bf16
                  const float*  __restrict__ biP,    // (5,160)
                  float* __restrict__ cNext,
                  float* __restrict__ hNext,
                  int M) {
    const int lane  = threadIdx.x & 31;
    const int wave  = threadIdx.x >> 5;
    const int mTile = blockIdx.x * 8 + wave;
    if (mTile * 16 >= M) return;                     // wave-uniform exit

    const int m0   = mTile * 16;
    const int l15  = lane & 15;
    const int half = lane >> 4;
    const int khA  = half * 8;
    const int khB  = half * 8;
    const int row  = m0 + l15;                       // may exceed M: slack rows
    const int col  = blockIdx.y * 16 + l15;

    const float* __restrict__ hl = hPrev + (size_t)(2 * row) * N_PAD;
    const float* __restrict__ hr = hl + N_PAD;

    v8f acc[5];
    #pragma unroll
    for (int g = 0; g < 5; ++g) acc[g] = (v8f){};

    #pragma unroll
    for (int kb = 0; kb < N_PAD; kb += 32) {
        const v16bf a = build_a_sum(hl, hr, kb + khA);
        #pragma unroll
        for (int g = 0; g < 5; ++g) {
            const v16bf b = load_b_bf16(
                WiP + ((size_t)(g * N_PAD + col)) * N_PAD + kb + khB);
            acc[g] = __builtin_amdgcn_wmma_f32_16x16x32_bf16(
                         false, a, false, b, (short)0, acc[g], false, false);
        }
    }

    const bool colOK = (col < MEM_DIM);
    const float bI  = 2.0f * biP[col];
    const float bO  = 2.0f * biP[N_PAD + col];
    const float bU  = 2.0f * biP[2 * N_PAD + col];
    const float bLF = 2.0f * biP[3 * N_PAD + col];
    const float bRF = 2.0f * biP[4 * N_PAD + col];

    #pragma unroll
    for (int r = 0; r < 8; ++r) {
        const int m = m0 + r + (half << 3);
        const float lc = cPrev[(size_t)(2 * m) * N_PAD + col];
        const float rc = cPrev[(size_t)(2 * m + 1) * N_PAD + col];
        const float cn = sigf(acc[0][r] + bI) * tanhf(acc[2][r] + bU)
                       + (acc[3][r] + bLF) * lc + (acc[4][r] + bRF) * rc;
        const float hn = sigf(acc[1][r] + bO) * tanhf(cn);
        cNext[(size_t)m * N_PAD + col] = colOK ? cn : 0.0f;
        hNext[(size_t)m * N_PAD + col] = colOK ? hn : 0.0f;
    }
}

// ---------------------------------------------------------------------------
// Persistent tail: levels M = 32,16,8,4,2,1 in ONE workgroup (16 waves),
// c/h ping-ponged through LDS, __syncthreads() between levels, root -> d_out.
// Wave->tile map is wave-uniform so EXEC is all-ones around every WMMA.
// ---------------------------------------------------------------------------
__global__ void __launch_bounds__(512)
tail_kernel(const float* __restrict__ cPrevG,  // (64+, 160) from last grid level
            const float* __restrict__ hPrevG,
            const __bf16* __restrict__ WiP,
            const float*  __restrict__ biP,
            float* __restrict__ out) {          // 300 floats: c then h
    __shared__ float c0[TAIL_M * N_PAD];        // 32x160
    __shared__ float h0[TAIL_M * N_PAD];
    __shared__ float c1[16 * N_PAD];            // 16x160
    __shared__ float h1[16 * N_PAD];

    const int lane = threadIdx.x & 31;
    const int wv   = threadIdx.x >> 5;          // 0..15
    const int l15  = lane & 15;
    const int half = lane >> 4;
    const int khA  = half * 8;
    const int khB  = half * 8;

    int level = 0;
    for (int M = TAIL_M; M >= 1; M >>= 1, ++level) {
        const int  tilesM = (M + 15) / 16;      // 2 then 1,1,1,1,1
        const bool active = (wv < tilesM * 5);  // wave-uniform
        const bool fromG  = (level == 0);
        float* cD = (level & 1) ? c1 : c0;
        float* hD = (level & 1) ? h1 : h0;
        const float* cS = fromG ? cPrevG : ((level & 1) ? c0 : c1);
        const float* hS = fromG ? hPrevG : ((level & 1) ? h0 : h1);

        if (active) {
            const int mTile = wv / 5;
            const int cp    = wv % 5;           // column pair: cols cp*32..+31
            const int m0    = mTile * 16;
            const int row   = m0 + l15;

            const float* hl = hS + (size_t)(2 * row) * N_PAD;
            const float* hr = hl + N_PAD;

            v8f acc[2][5];
            #pragma unroll
            for (int ct = 0; ct < 2; ++ct)
                #pragma unroll
                for (int g = 0; g < 5; ++g) acc[ct][g] = (v8f){};

            #pragma unroll
            for (int kb = 0; kb < N_PAD; kb += 32) {
                const v16bf a = build_a_sum(hl, hr, kb + khA);
                #pragma unroll
                for (int ct = 0; ct < 2; ++ct) {
                    const int col = cp * 32 + ct * 16 + l15;
                    #pragma unroll
                    for (int g = 0; g < 5; ++g) {
                        const v16bf b = load_b_bf16(
                            WiP + ((size_t)(g * N_PAD + col)) * N_PAD + kb + khB);
                        acc[ct][g] = __builtin_amdgcn_wmma_f32_16x16x32_bf16(
                            false, a, false, b, (short)0, acc[ct][g], false, false);
                    }
                }
            }

            #pragma unroll
            for (int ct = 0; ct < 2; ++ct) {
                const int  col   = cp * 32 + ct * 16 + l15;
                const bool colOK = (col < MEM_DIM);
                const float bI  = 2.0f * biP[col];
                const float bO  = 2.0f * biP[N_PAD + col];
                const float bU  = 2.0f * biP[2 * N_PAD + col];
                const float bLF = 2.0f * biP[3 * N_PAD + col];
                const float bRF = 2.0f * biP[4 * N_PAD + col];
                #pragma unroll
                for (int r = 0; r < 8; ++r) {
                    const int m = m0 + r + (half << 3);
                    const float lc = cS[(size_t)(2 * m) * N_PAD + col];
                    const float rc = cS[(size_t)(2 * m + 1) * N_PAD + col];
                    const float cn = sigf(acc[ct][0][r] + bI) * tanhf(acc[ct][2][r] + bU)
                                   + (acc[ct][3][r] + bLF) * lc
                                   + (acc[ct][4][r] + bRF) * rc;
                    const float hn = sigf(acc[ct][1][r] + bO) * tanhf(cn);
                    if (M == 1) {
                        if (m == 0 && colOK) {
                            out[col]           = cn;
                            out[MEM_DIM + col] = hn;
                        }
                    } else {
                        cD[m * N_PAD + col] = colOK ? cn : 0.0f;
                        hD[m * N_PAD + col] = colOK ? hn : 0.0f;
                    }
                }
            }
        }
        __syncthreads();
    }
}

// ---------------------------------------------------------------------------
static inline size_t alignUp(size_t v, size_t a) { return (v + a - 1) & ~(a - 1); }

extern "C" void kernel_launch(void* const* d_in, const int* in_sizes, int n_in,
                              void* d_out, int out_size, void* d_ws, size_t ws_size,
                              hipStream_t stream) {
    (void)in_sizes; (void)n_in; (void)out_size; (void)ws_size;
    const float* X  = (const float*)d_in[0];   // (131072, 300)
    const float* Wl = (const float*)d_in[1];   // (150, 300)
    const float* bl = (const float*)d_in[2];   // (150,)
    const float* Wi = (const float*)d_in[3];   // (750, 150)
    const float* bi = (const float*)d_in[4];   // (750,)
    float* out = (float*)d_out;                // 300 floats: c then h

    // Workspace carve-up (256B aligned).
    char* ws = (char*)d_ws;
    size_t off = 0;
    __bf16* WlP = (__bf16*)(ws + off); off = alignUp(off + (size_t)N_PAD * K_XPAD * 2, 256);
    __bf16* WiP = (__bf16*)(ws + off); off = alignUp(off + (size_t)5 * N_PAD * N_PAD * 2, 256);
    float*  blP = (float*)(ws + off);  off = alignUp(off + (size_t)N_PAD * 4, 256);
    float*  biP = (float*)(ws + off);  off = alignUp(off + (size_t)5 * N_PAD * 4, 256);
    const size_t rowsA = (size_t)N_LEAVES + ROW_SLACK;
    const size_t rowsB = (size_t)(N_LEAVES / 2) + ROW_SLACK;
    float* cA = (float*)(ws + off); off += rowsA * N_PAD * sizeof(float);
    float* hA = (float*)(ws + off); off += rowsA * N_PAD * sizeof(float);
    float* cB = (float*)(ws + off); off += rowsB * N_PAD * sizeof(float);
    float* hB = (float*)(ws + off); off += rowsB * N_PAD * sizeof(float);

    // 1) Weight/bias padding + bf16 conversion.
    {
        int n = 5 * N_PAD * N_PAD;   // 128000 covers every sub-range
        prep_weights_kernel<<<(n + 255) / 256, 256, 0, stream>>>(
            Wl, bl, Wi, bi, WlP, blP, WiP, biP);
    }
    // 2) Leaf GEMM + activation (fused fp32->bf16 conversion of X).
    leaf_kernel<<<dim3(N_LEAVES / (16 * 8)), 256, 0, stream>>>(X, WlP, blP, cA, hA);

    // 3) Grid-wide tree levels down to M=64, ping-ponging A <-> B.
    float* cCur = cA; float* hCur = hA;
    bool toB = true;
    int n = N_LEAVES;
    while (n > 2 * TAIL_M) {                         // levels M=65536..64
        const int M = n / 2;
        float* cN = toB ? cB : cA;
        float* hN = toB ? hB : hA;
        dim3 grid((M + 127) / 128, N_PAD / 16);
        tree_level_kernel<<<grid, 256, 0, stream>>>(cCur, hCur, WiP, biP, cN, hN, M);
        cCur = cN; hCur = hN;
        toB = !toB;
        n = M;
    }
    // 4) Persistent tail: M = 32..1 inside one workgroup via LDS; writes d_out.
    tail_kernel<<<1, 512, 0, stream>>>(cCur, hCur, WiP, biP, out);
}
